// WindowAttention_21784074125560
// MI455X (gfx1250) — matrix operand order repaired and checked by reading, not compiled, not run on hardware
//
#include <hip/hip_runtime.h>
#include <hip/hip_bf16.h>
#include <stdint.h>

#define HEADS 16
#define HDIM  32
#define NTOK  49
#define NPAD  64
#define DIM   512
#define SCALE 0.17677669529663687f   // 32^-0.5

typedef __attribute__((ext_vector_type(16))) __bf16 bf16x16;
typedef __attribute__((ext_vector_type(8)))  __bf16 bf16x8;
typedef __attribute__((ext_vector_type(2)))  __bf16 bf16x2;
typedef __attribute__((ext_vector_type(8)))  float  f32x8;
typedef __attribute__((ext_vector_type(2)))  float  f32x2;

// Native f32 -> bf16 (RNE).
static __device__ __forceinline__ unsigned short f2bf(float f) {
  __bf16 h = (__bf16)f;
  return __builtin_bit_cast(unsigned short, h);
}
// Two floats -> packed bf16 dword, straight out of v_cvt_pk_bf16_f32.
static __device__ __forceinline__ unsigned pk2(float a, float b) {
  f32x2 v; v.x = a; v.y = b;
  bf16x2 c = __builtin_convertvector(v, bf16x2);
  return __builtin_bit_cast(unsigned, c);
}
// float4 -> 4 packed bf16 (uint2).
static __device__ __forceinline__ uint2 pk4(float4 v) {
  uint2 u; u.x = pk2(v.x, v.y); u.y = pk2(v.z, v.w);
  return u;
}

// A-fragment: 16(M) x 32(K) bf16 from row-major [16][stride] u16.
// lane l: row = l&15; K chunks {8*(l>>4)..+7} and {16+8*(l>>4)..+7}.
static __device__ __forceinline__ bf16x16 frag_a(const unsigned short* base, int stride) {
  const int lane = threadIdx.x & 31;
  const unsigned short* p = base + (lane & 15) * stride + ((lane >> 4) << 3);
  bf16x8 lo = *reinterpret_cast<const bf16x8*>(p);
  bf16x8 hi = *reinterpret_cast<const bf16x8*>(p + 16);
  return __builtin_shufflevector(lo, hi, 0,1,2,3,4,5,6,7,8,9,10,11,12,13,14,15);
}

// B-fragment: 32(K) x 16(N) bf16 from row-major B^T [16N][stride] u16.
// lane l: col n = l&15; K chunks {8*(l>>4)..+7} and {16+8*(l>>4)..+7}.
static __device__ __forceinline__ bf16x16 frag_b(const unsigned short* base, int stride) {
  const int lane = threadIdx.x & 31;
  const unsigned short* p = base + (lane & 15) * stride + ((lane >> 4) << 3);
  bf16x8 lo = *reinterpret_cast<const bf16x8*>(p);
  bf16x8 hi = *reinterpret_cast<const bf16x8*>(p + 16);
  return __builtin_shufflevector(lo, hi, 0,1,2,3,4,5,6,7,8,9,10,11,12,13,14,15);
}

static __device__ __forceinline__ f32x8 wmma_bf16(bf16x16 a, bf16x16 b, f32x8 c) {
  return __builtin_amdgcn_wmma_f32_16x16x32_bf16(false, a, false, b, (short)0, c, false, false);
}

// ---------------------------------------------------------------------------
// Kernel 1: C = x @ qkv_w^T + qkv_b, scatter to Q(scaled)/K [b][h][64][32]
// and V^T [b][h][32][64], all bf16.  Block tile 64(M) x 128(N), 8 waves,
// double-buffered LDS (one barrier / K-step), load/convert phases split.
// ---------------------------------------------------------------------------
__global__ __launch_bounds__(256) void qkv_gemm_kernel(
    const float* __restrict__ x, const float* __restrict__ w,
    const float* __restrict__ bias,
    unsigned short* __restrict__ Qs, unsigned short* __restrict__ Ks,
    unsigned short* __restrict__ Vt)
{
  __shared__ unsigned short As[2][64 * 40];   // stride 40 u16 = 80B: conflict-free
  __shared__ unsigned short Bs[2][128 * 40];
  const int t = threadIdx.x;
  const int lane = t & 31, wave = t >> 5;
  const int wm = wave >> 2, wn = wave & 3;           // 2 x 4 wave grid
  const int nb = blockIdx.x % 12;
  const int mb = blockIdx.x / 12;
  const int mBase = mb * 64, nBase = nb * 128;

  f32x8 accs[2][2] = {};

  auto stage = [&](int kk, int bi) {
    const int k0 = kk << 5;
    float4 va[2], vb[4];
    // phase 1: issue all global loads
#pragma unroll
    for (int i = 0; i < 2; ++i) {             // A: 64x32 f32
      int id = t + (i << 8);
      int row = id >> 3, c4 = (id & 7) << 2;
      const float* src = x + (size_t)(mBase + row) * DIM + k0 + c4;
      if (kk < 15) __builtin_prefetch(src + 32, 0, 3);
      va[i] = *reinterpret_cast<const float4*>(src);
    }
#pragma unroll
    for (int i = 0; i < 4; ++i) {             // B: 128x32 f32
      int id = t + (i << 8);
      int row = id >> 3, c4 = (id & 7) << 2;
      const float* src = w + (size_t)(nBase + row) * DIM + k0 + c4;
      if (kk < 15) __builtin_prefetch(src + 32, 0, 3);
      vb[i] = *reinterpret_cast<const float4*>(src);
    }
    // phase 2: convert + LDS store
#pragma unroll
    for (int i = 0; i < 2; ++i) {
      int id = t + (i << 8);
      int row = id >> 3, c4 = (id & 7) << 2;
      *reinterpret_cast<uint2*>(&As[bi][row * 40 + c4]) = pk4(va[i]);
    }
#pragma unroll
    for (int i = 0; i < 4; ++i) {
      int id = t + (i << 8);
      int row = id >> 3, c4 = (id & 7) << 2;
      *reinterpret_cast<uint2*>(&Bs[bi][row * 40 + c4]) = pk4(vb[i]);
    }
  };

  stage(0, 0);
  __syncthreads();
  for (int kk = 0; kk < 16; ++kk) {
    const int cur = kk & 1;
    if (kk < 15) stage(kk + 1, cur ^ 1);
    bf16x16 a0 = frag_a(&As[cur][(wm * 32     ) * 40], 40);
    bf16x16 a1 = frag_a(&As[cur][(wm * 32 + 16) * 40], 40);
    bf16x16 b0 = frag_b(&Bs[cur][(wn * 32     ) * 40], 40);
    bf16x16 b1 = frag_b(&Bs[cur][(wn * 32 + 16) * 40], 40);
    accs[0][0] = wmma_bf16(a0, b0, accs[0][0]);
    accs[0][1] = wmma_bf16(a0, b1, accs[0][1]);
    accs[1][0] = wmma_bf16(a1, b0, accs[1][0]);
    accs[1][1] = wmma_bf16(a1, b1, accs[1][1]);
    __syncthreads();
  }

  const int rOff = (lane >> 4) << 3;   // C row = i + 8*(lane>>4)
  const int cOff = lane & 15;          // C col = lane&15
#pragma unroll
  for (int ti = 0; ti < 2; ++ti)
#pragma unroll
    for (int tj = 0; tj < 2; ++tj) {
      const int n = nBase + wn * 32 + tj * 16 + cOff;     // 0..1535
      const int which = n / DIM;                          // 0=q 1=k 2=v
      const int c = n & (DIM - 1);
      const int h = c >> 5, dd = c & 31;
      const float bn = bias[n];
#pragma unroll
      for (int i = 0; i < 8; ++i) {
        const int m = mBase + wm * 32 + ti * 16 + rOff + i;
        const float val = accs[ti][tj][i] + bn;
        const int b = m / NTOK, tok = m - b * NTOK;
        const size_t bh = (size_t)b * HEADS + h;
        if (which == 0)      Qs[(bh * NPAD + tok) * HDIM + dd] = f2bf(val * SCALE);
        else if (which == 1) Ks[(bh * NPAD + tok) * HDIM + dd] = f2bf(val);
        else                 Vt[(bh * HDIM + dd) * NPAD + tok] = f2bf(val);
      }
    }
}

// ---------------------------------------------------------------------------
// Kernel 2: per (window, head) attention. 1 wave per head, 2 waves/block.
// S = qk^T (+bias +mask), fp32 softmax in LDS, O = P @ V via WMMA.
// ---------------------------------------------------------------------------
__global__ __launch_bounds__(64) void attn_kernel(
    const unsigned short* __restrict__ Qs, const unsigned short* __restrict__ Ks,
    const unsigned short* __restrict__ Vt,
    const float* __restrict__ mask, const float* __restrict__ rpb,
    unsigned short* __restrict__ AO, int nW)
{
  __shared__ float          Ssh[2][64 * 68];   // stride 68 f32 = 272B (16B mult)
  __shared__ unsigned short Psh[2][64 * 72];   // stride 72 u16 = 144B (16B mult)
  const int t = threadIdx.x, lane = t & 31, wave = t >> 5;
  const int b = blockIdx.x >> 3;
  const int h = ((blockIdx.x & 7) << 1) + wave;
  const size_t bh = (size_t)b * HEADS + h;
  const unsigned short* Qb = Qs + bh * (NPAD * HDIM);
  const unsigned short* Kb = Ks + bh * (NPAD * HDIM);
  const unsigned short* Vb = Vt + bh * (HDIM * NPAD);
  const float* mw = mask + (size_t)(b % nW) * (NTOK * NTOK);
  float* S = Ssh[wave];
  unsigned short* P = Psh[wave];

  bf16x16 qf[4], kf[4];
#pragma unroll
  for (int i = 0; i < 4; ++i) {
    qf[i] = frag_a(Qb + i * 16 * HDIM, HDIM);
    kf[i] = frag_b(Kb + i * 16 * HDIM, HDIM);
  }

  const int rOff = (lane >> 4) << 3;
  const int cOff = lane & 15;

  // hoist per-column rel-pos terms (col depends only on ni)
  int colv[4], jhv[4], jwv[4];
#pragma unroll
  for (int ni = 0; ni < 4; ++ni) {
    colv[ni] = ni * 16 + cOff;
    jhv[ni] = colv[ni] / 7;
    jwv[ni] = colv[ni] - jhv[ni] * 7;
  }

#pragma unroll
  for (int mi = 0; mi < 4; ++mi) {
    // hoist per-row rel-pos terms (row depends only on mi, i)
    int rowv[8], ihv[8], iwv[8];
#pragma unroll
    for (int i = 0; i < 8; ++i) {
      rowv[i] = mi * 16 + rOff + i;
      ihv[i] = rowv[i] / 7;
      iwv[i] = rowv[i] - ihv[i] * 7;
    }
#pragma unroll
    for (int ni = 0; ni < 4; ++ni) {
      f32x8 acc = {};
      acc = wmma_bf16(qf[mi], kf[ni], acc);     // K = head_dim = 32, one step
      const int col = colv[ni];
#pragma unroll
      for (int i = 0; i < 8; ++i) {
        const int row = rowv[i];
        float v = -1e30f;
        if (row < NTOK && col < NTOK) {
          const int idx = (ihv[i] - jhv[ni] + 6) * 13 + (iwv[i] - jwv[ni] + 6);
          v = acc[i] + rpb[(idx << 4) + h] + mw[row * NTOK + col];
        }
        S[row * 68 + col] = v;
      }
    }
  }
  __syncthreads();

  // fp32 softmax, one lane per row (2 rows/lane); emit P as packed bf16 pairs
#pragma unroll
  for (int rr = 0; rr < 2; ++rr) {
    const int r = lane + (rr << 5);
    float* row = S + r * 68;
    float mx = -1e30f;
    for (int j = 0; j < NTOK; ++j) mx = fmaxf(mx, row[j]);
    float sum = 0.f;
    for (int j = 0; j < NTOK; ++j) { float e = __expf(row[j] - mx); sum += e; row[j] = e; }
    const float inv = 1.f / sum;
    unsigned* pr = reinterpret_cast<unsigned*>(P + r * 72);
#pragma unroll 8
    for (int j2 = 0; j2 < NPAD / 2; ++j2) {
      const int j0 = 2 * j2, j1 = 2 * j2 + 1;
      const float p0 = (j0 < NTOK) ? row[j0] * inv : 0.f;
      const float p1 = (j1 < NTOK) ? row[j1] * inv : 0.f;
      pr[j2] = pk2(p0, p1);
    }
  }
  __syncthreads();

  // O = P(64x64) @ V(64x32): 4 M-tiles x 2 N-tiles x 2 K-steps
#pragma unroll
  for (int mi = 0; mi < 4; ++mi) {
#pragma unroll
    for (int nj = 0; nj < 2; ++nj) {
      f32x8 acc = {};
#pragma unroll
      for (int kt = 0; kt < 2; ++kt) {
        bf16x16 a  = frag_a(P  + (mi * 16) * 72   + kt * 32, 72);
        bf16x16 bb = frag_b(Vb + (nj * 16) * NPAD + kt * 32, NPAD);
        acc = wmma_bf16(a, bb, acc);
      }
      const int d = nj * 16 + cOff;
#pragma unroll
      for (int i = 0; i < 8; ++i) {
        const int tok = mi * 16 + rOff + i;
        if (tok < NTOK)
          AO[((size_t)b * NTOK + tok) * DIM + h * HDIM + d] = f2bf(acc[i]);
      }
    }
  }
}

// ---------------------------------------------------------------------------
// Kernel 3: out = AO(bf16) @ proj_w^T + proj_b (f32 out).  Tile 64x128,
// double-buffered; A tile staged with async global->LDS (CDNA5), B converted.
// ---------------------------------------------------------------------------
__global__ __launch_bounds__(256) void proj_gemm_kernel(
    const unsigned short* __restrict__ A, const float* __restrict__ w,
    const float* __restrict__ bias, float* __restrict__ out)
{
  __shared__ unsigned short As[2][64 * 40];
  __shared__ unsigned short Bs[2][128 * 40];
  const int t = threadIdx.x;
  const int lane = t & 31, wave = t >> 5;
  const int wm = wave >> 2, wn = wave & 3;
  const int nb = blockIdx.x & 3;
  const int mb = blockIdx.x >> 2;
  const int mBase = mb * 64, nBase = nb * 128;

  f32x8 accs[2][2] = {};

  // A tile (64x32 bf16 = 256 x 16B chunks, one per thread): async copy to LDS.
  auto stageA = [&](int kk, int bi) {
    const int row = t >> 2, ch = t & 3;             // 64 rows x 4 chunks
    const unsigned short* src = A + (size_t)(mBase + row) * DIM + (kk << 5) + (ch << 3);
    const unsigned lds = (unsigned)(size_t)&As[bi][row * 40 + (ch << 3)];
    const unsigned long long ga = (unsigned long long)(size_t)src;
    asm volatile("global_load_async_to_lds_b128 %0, %1, off"
                 :: "v"(lds), "v"(ga) : "memory");
  };
  auto stageB = [&](int kk, int bi) {
    const int k0 = kk << 5;
    float4 vb[4];
#pragma unroll
    for (int i = 0; i < 4; ++i) {
      int id = t + (i << 8);
      int row = id >> 3, c4 = (id & 7) << 2;
      const float* src = w + (size_t)(nBase + row) * DIM + k0 + c4;
      if (kk < 15) __builtin_prefetch(src + 32, 0, 3);
      vb[i] = *reinterpret_cast<const float4*>(src);
    }
#pragma unroll
    for (int i = 0; i < 4; ++i) {
      int id = t + (i << 8);
      int row = id >> 3, c4 = (id & 7) << 2;
      *reinterpret_cast<uint2*>(&Bs[bi][row * 40 + c4]) = pk4(vb[i]);
    }
  };

  stageA(0, 0);
  stageB(0, 0);
  asm volatile("s_wait_asynccnt 0x0" ::: "memory");
  __syncthreads();
  for (int kk = 0; kk < 16; ++kk) {
    const int cur = kk & 1;
    if (kk < 15) { stageA(kk + 1, cur ^ 1); stageB(kk + 1, cur ^ 1); }
    bf16x16 a0 = frag_a(&As[cur][(wm * 32     ) * 40], 40);
    bf16x16 a1 = frag_a(&As[cur][(wm * 32 + 16) * 40], 40);
    bf16x16 b0 = frag_b(&Bs[cur][(wn * 32     ) * 40], 40);
    bf16x16 b1 = frag_b(&Bs[cur][(wn * 32 + 16) * 40], 40);
    accs[0][0] = wmma_bf16(a0, b0, accs[0][0]);
    accs[0][1] = wmma_bf16(a0, b1, accs[0][1]);
    accs[1][0] = wmma_bf16(a1, b0, accs[1][0]);
    accs[1][1] = wmma_bf16(a1, b1, accs[1][1]);
    asm volatile("s_wait_asynccnt 0x0" ::: "memory");
    __syncthreads();
  }

  const int rOff = (lane >> 4) << 3;
  const int cOff = lane & 15;
#pragma unroll
  for (int ti = 0; ti < 2; ++ti)
#pragma unroll
    for (int tj = 0; tj < 2; ++tj) {
      const int n = nBase + wn * 32 + tj * 16 + cOff;
      const float bn = bias[n];
#pragma unroll
      for (int i = 0; i < 8; ++i) {
        const int m = mBase + wm * 32 + ti * 16 + rOff + i;
        out[(size_t)m * DIM + n] = accs[ti][tj][i] + bn;
      }
    }
}

// ---------------------------------------------------------------------------
extern "C" void kernel_launch(void* const* d_in, const int* in_sizes, int n_in,
                              void* d_out, int out_size, void* d_ws, size_t ws_size,
                              hipStream_t stream) {
  const float* x      = (const float*)d_in[0];
  const float* mask   = (const float*)d_in[1];
  const float* qkv_w  = (const float*)d_in[2];
  const float* qkv_b  = (const float*)d_in[3];
  const float* proj_w = (const float*)d_in[4];
  const float* proj_b = (const float*)d_in[5];
  const float* rpb    = (const float*)d_in[6];

  const int B_ = in_sizes[0] / (NTOK * DIM);          // 2048
  const int nW = in_sizes[1] / (NTOK * NTOK);         // 64
  const int M  = B_ * NTOK;                           // 100352 (mult of 64)

  const size_t qkSz = (size_t)B_ * HEADS * NPAD * HDIM;  // elems (u16)
  unsigned short* Qs = (unsigned short*)d_ws;
  unsigned short* Ks = Qs + qkSz;
  unsigned short* Vt = Ks + qkSz;                     // [b][h][32][64]
  unsigned short* AO = Vt + qkSz;                     // [b][49][512] bf16

  // zero V^T (incl. token padding 49..63) so padded P columns hit finite zeros
  hipMemsetAsync(Vt, 0, qkSz * sizeof(unsigned short), stream);

  qkv_gemm_kernel<<<(M / 64) * 12, 256, 0, stream>>>(x, qkv_w, qkv_b, Qs, Ks, Vt);
  attn_kernel<<<B_ * 8, 64, 0, stream>>>(Qs, Ks, Vt, mask, rpb, AO, nW);
  proj_gemm_kernel<<<(M / 64) * 4, 256, 0, stream>>>(AO, proj_w, proj_b, (float*)d_out);
}
